// FinePreprocess_84507776516562
// MI455X (gfx1250) — matrix operand order repaired and checked by reading, not compiled, not run on hardware
//
#include <hip/hip_runtime.h>

// ---------------------------------------------------------------------------
// Types for CDNA5 WMMA (wave32, v_wmma_f32_16x16x32_bf16)
// ---------------------------------------------------------------------------
typedef __attribute__((ext_vector_type(16))) __bf16        v16bf;
typedef __attribute__((ext_vector_type(8)))  float         v8f;
typedef __attribute__((ext_vector_type(4)))  unsigned int  u32x4;

union Frag {
  v16bf bf;
  u32x4 u[2];
};

__device__ __forceinline__ unsigned short f2bf(float f) {
  unsigned int u = __float_as_uint(f);
  u += 0x7FFFu + ((u >> 16) & 1u);        // round-to-nearest-even
  return (unsigned short)(u >> 16);
}

__device__ __forceinline__ v16bf load_frag2(const unsigned short* p0,
                                            const unsigned short* p1) {
  Frag f;
  f.u[0] = *reinterpret_cast<const u32x4*>(p0);
  f.u[1] = *reinterpret_cast<const u32x4*>(p1);
  return f.bf;
}

// Problem constants (match setup_inputs)
#define HF 240
#define WFW 320
#define CF 128
#define LC 4800
#define CC 256

// ---------------------------------------------------------------------------
// prep_comb: W_comb[n][k] = sum_j W_merge[n][128+j] * W_down[j][k]  (fp32)
//            b_comb[n]    = b_merge[n] + sum_j W_merge[n][128+j] * b_down[j]
// grid <<<128, 256>>> : block n, thread k
// ---------------------------------------------------------------------------
__global__ void prep_comb(const float* __restrict__ W_down,
                          const float* __restrict__ b_down,
                          const float* __restrict__ W_merge,
                          const float* __restrict__ b_merge,
                          float* __restrict__ wcomb,
                          float* __restrict__ bcomb) {
  int n = blockIdx.x;
  int k = threadIdx.x;
  const float* wm2 = W_merge + n * 256 + 128;
  float s = 0.f;
  for (int j = 0; j < 128; ++j) s += wm2[j] * W_down[j * 256 + k];
  wcomb[n * 256 + k] = s;
  if (k == 0) {
    float sb = b_merge[n];
    for (int j = 0; j < 128; ++j) sb += wm2[j] * b_down[j];
    bcomb[n] = sb;
  }
}

// ---------------------------------------------------------------------------
// prep_pack: pack B matrices into per-lane WMMA fragment order (bf16).
// B order used here (contiguous k, same as the A loads; not the ISA table order): lane = N (lanes 16-31 -> K+16),
// halves i=0..15 -> K = koff + i with koff = (lane&16)?16:0.
// bpackM [kblk4][nblk8][lane32][16]  from W_merge[:, :128]   (16384 elems)
// bpackC [kblk8][nblk8][lane32][16]  from wcomb[128][256]    (32768 elems)
// grid <<<192, 256>>>
// ---------------------------------------------------------------------------
__global__ void prep_pack(const float* __restrict__ W_merge,
                          const float* __restrict__ wcomb,
                          unsigned short* __restrict__ bpackM,
                          unsigned short* __restrict__ bpackC) {
  int idx = blockIdx.x * blockDim.x + threadIdx.x;
  if (idx < 16384) {
    int i = idx & 15, lane = (idx >> 4) & 31, nblk = (idx >> 9) & 7, kblk = idx >> 12;
    int k = kblk * 32 + ((lane & 16) ? 16 : 0) + i;       // k < 128
    int n = nblk * 16 + (lane & 15);
    bpackM[idx] = f2bf(W_merge[n * 256 + k]);             // Wm1[n][k]
  } else if (idx < 16384 + 32768) {
    int id = idx - 16384;
    int i = id & 15, lane = (id >> 4) & 31, nblk = (id >> 9) & 7, kblk = id >> 12;
    int k = kblk * 32 + ((lane & 16) ? 16 : 0) + i;       // k < 256
    int n = nblk * 16 + (lane & 15);
    bpackC[id] = f2bf(wcomb[n * 256 + k]);
  }
}

// ---------------------------------------------------------------------------
// coarse_kernel: t[r][n] = c_cat[r][:] @ wcomb[n][:] + bcomb[n]
// c_cat[r] = r<M ? feat_c0[b_ids[r], i_ids[r]] : feat_c1[b_ids[r-M], j_ids[r-M]]
// One WG = 32 rows; 8 waves, wave w -> N tile w; 8 K-steps x 2 M-tiles WMMA.
// ---------------------------------------------------------------------------
__global__ void __launch_bounds__(256)
coarse_kernel(const float* __restrict__ feat_c0,
              const float* __restrict__ feat_c1,
              const int* __restrict__ b_ids,
              const int* __restrict__ i_ids,
              const int* __restrict__ j_ids,
              const unsigned short* __restrict__ bpackC,
              const float* __restrict__ bcomb,
              float* __restrict__ tbuf,
              int M) {
  __shared__ __align__(16) unsigned short Cbuf[32][CC + 8];  // +16B row pad
  const int R = 2 * M;
  const int r0 = blockIdx.x * 32;
  const int tid = threadIdx.x;

  // Gather 32 x 256 coarse descriptors -> LDS bf16 (coalesced per row)
  for (int e = tid; e < 32 * 256; e += 256) {
    int row = e >> 8, k = e & 255;
    int r = r0 + row;
    float v = 0.f;
    if (r < R) {
      int side = (r >= M);
      int mi = side ? r - M : r;
      int b = b_ids[mi];
      int loc = side ? j_ids[mi] : i_ids[mi];
      const float* src = side ? feat_c1 : feat_c0;
      v = src[((size_t)b * LC + loc) * CC + k];
    }
    Cbuf[row][k] = f2bf(v);
  }
  __syncthreads();

  const int wave = tid >> 5, lane = tid & 31;
  const int nblk = wave;
  const int n = nblk * 16 + (lane & 15);
  const float bv = bcomb[n];
  v8f acc0, acc1;
#pragma unroll
  for (int q = 0; q < 8; ++q) { acc0[q] = bv; acc1[q] = bv; }

  const int kbaseA = (lane & 16) ? 16 : 0;
  const unsigned short* arow0 = &Cbuf[lane & 15][0];
  const unsigned short* arow1 = &Cbuf[16 + (lane & 15)][0];
#pragma unroll
  for (int kb = 0; kb < 8; ++kb) {
    int ka = kb * 32 + kbaseA;
    v16bf a0 = load_frag2(arow0 + ka, arow0 + ka + 8);
    v16bf a1 = load_frag2(arow1 + ka, arow1 + ka + 8);
    const unsigned short* bp = bpackC + (((kb * 8 + nblk) * 32 + lane) << 4);
    v16bf bf = load_frag2(bp, bp + 8);  // 32 contiguous bytes per lane
    acc0 = __builtin_amdgcn_wmma_f32_16x16x32_bf16(false, a0, false, bf, (short)0, acc0, false, false);
    acc1 = __builtin_amdgcn_wmma_f32_16x16x32_bf16(false, a1, false, bf, (short)0, acc1, false, false);
  }

  const int rlo = (lane & 16) ? 8 : 0;
#pragma unroll
  for (int q = 0; q < 8; ++q) {
    int ra = r0 + q + rlo;          // M-tile 0
    int rb = r0 + 16 + q + rlo;     // M-tile 1
    if (ra < R) tbuf[(size_t)ra * 128 + n] = acc0[q];
    if (rb < R) tbuf[(size_t)rb * 128 + n] = acc1[q];
  }
}

// ---------------------------------------------------------------------------
// merge_kernel: one WG per row r in [0, 2M).
// out[r][p][n] = window[p][:] @ Wm1[n][:] + t[r][n],  p in [0,25)
// Window gather 25x128 -> LDS bf16 (rows padded to 32), then 8 waves x
// (4 K-steps x 2 M-tiles) v_wmma_f32_16x16x32_bf16.
// ---------------------------------------------------------------------------
__global__ void __launch_bounds__(256)
merge_kernel(const float* __restrict__ feat_f0,
             const float* __restrict__ feat_f1,
             const int* __restrict__ b_ids,
             const int* __restrict__ i_ids,
             const int* __restrict__ j_ids,
             const int* __restrict__ stride_p,
             const unsigned short* __restrict__ bpackM,
             const float* __restrict__ tbuf,
             float* __restrict__ out,
             int M) {
  __shared__ __align__(16) unsigned short Abuf[32][CF + 8];  // +16B row pad

  const int r = blockIdx.x;
  const int side = (r >= M);
  const int mi = side ? r - M : r;
  const int b = b_ids[mi];
  const int loc = side ? j_ids[mi] : i_ids[mi];
  const int stride = stride_p[0];
  const int Wc = (WFW + 4 - 5) / stride + 1;      // 80 for stride 4
  const int y = loc / Wc, x = loc % Wc;
  const float* fb = (side ? feat_f1 : feat_f0) + (size_t)b * CF * HF * WFW;
  const int tid = threadIdx.x;

  // Gather 5x5 window, all 128 channels, with zero padding at borders.
  // e = c*25 + p keeps 5-float row runs contiguous (L2-resident gather).
  for (int e = tid; e < 25 * CF; e += 256) {
    int c = e / 25, p = e - c * 25;
    int ky = p / 5, kx = p - ky * 5;
    int ry = y * stride + ky - 2;
    int rx = x * stride + kx - 2;
    float v = 0.f;
    if (ry >= 0 && ry < HF && rx >= 0 && rx < WFW)
      v = fb[(size_t)c * (HF * WFW) + ry * WFW + rx];
    Abuf[p][c] = f2bf(v);
  }
  for (int e = 25 * CF + tid; e < 32 * CF; e += 256)
    Abuf[e >> 7][e & 127] = 0;                    // zero rows 25..31
  __syncthreads();

  const int wave = tid >> 5, lane = tid & 31;
  const int nblk = wave;
  const int n = nblk * 16 + (lane & 15);
  const float tv = tbuf[(size_t)r * 128 + n];     // coarse contribution + bias
  v8f acc0, acc1;
#pragma unroll
  for (int q = 0; q < 8; ++q) { acc0[q] = tv; acc1[q] = tv; }

  const int kbaseA = (lane & 16) ? 16 : 0;
  const unsigned short* arow0 = &Abuf[lane & 15][0];
  const unsigned short* arow1 = &Abuf[16 + (lane & 15)][0];
#pragma unroll
  for (int kb = 0; kb < 4; ++kb) {
    int ka = kb * 32 + kbaseA;
    v16bf a0 = load_frag2(arow0 + ka, arow0 + ka + 8);
    v16bf a1 = load_frag2(arow1 + ka, arow1 + ka + 8);
    const unsigned short* bp = bpackM + (((kb * 8 + nblk) * 32 + lane) << 4);
    v16bf bf = load_frag2(bp, bp + 8);
    acc0 = __builtin_amdgcn_wmma_f32_16x16x32_bf16(false, a0, false, bf, (short)0, acc0, false, false);
    acc1 = __builtin_amdgcn_wmma_f32_16x16x32_bf16(false, a1, false, bf, (short)0, acc1, false, false);
  }

  float* outr = out + (size_t)r * (25 * 128);
  const int rlo = (lane & 16) ? 8 : 0;
#pragma unroll
  for (int q = 0; q < 8; ++q) {
    int p0 = q + rlo;                 // rows 0..15: always valid
    int p1 = 16 + q + rlo;            // rows 16..31: valid if < 25
    outr[p0 * 128 + n] = acc0[q];
    if (p1 < 25) outr[p1 * 128 + n] = acc1[q];
  }
}

// ---------------------------------------------------------------------------
// launch
// ---------------------------------------------------------------------------
extern "C" void kernel_launch(void* const* d_in, const int* in_sizes, int n_in,
                              void* d_out, int out_size, void* d_ws, size_t ws_size,
                              hipStream_t stream) {
  const float* feat_f0 = (const float*)d_in[0];
  const float* feat_f1 = (const float*)d_in[1];
  const float* feat_c0 = (const float*)d_in[2];
  const float* feat_c1 = (const float*)d_in[3];
  const int*   b_ids   = (const int*)d_in[4];
  const int*   i_ids   = (const int*)d_in[5];
  const int*   j_ids   = (const int*)d_in[6];
  const float* W_down  = (const float*)d_in[7];
  const float* b_down  = (const float*)d_in[8];
  const float* W_merge = (const float*)d_in[9];
  const float* b_merge = (const float*)d_in[10];
  const int*   stride_p = (const int*)d_in[11];
  const int M = in_sizes[4];
  float* out = (float*)d_out;

  // Workspace layout (bytes):
  //   [0, 32768)        bpackM  (bf16 fragments of W_merge[:, :128])
  //   [32768, 98304)    bpackC  (bf16 fragments of W_comb)
  //   [98304, 98816)    b_comb  fp32[128]
  //   [98816, 229888)   W_comb  fp32[128][256]
  //   [229888, ...)     t       fp32[2M][128]
  unsigned short* bpackM = (unsigned short*)d_ws;
  unsigned short* bpackC = (unsigned short*)((char*)d_ws + 32768);
  float* bcomb = (float*)((char*)d_ws + 98304);
  float* wcomb = (float*)((char*)d_ws + 98816);
  float* tbuf  = (float*)((char*)d_ws + 229888);

  prep_comb<<<128, 256, 0, stream>>>(W_down, b_down, W_merge, b_merge, wcomb, bcomb);
  prep_pack<<<192, 256, 0, stream>>>(W_merge, wcomb, bpackM, bpackC);

  const int R = 2 * M;
  coarse_kernel<<<(R + 31) / 32, 256, 0, stream>>>(feat_c0, feat_c1, b_ids, i_ids,
                                                   j_ids, bpackC, bcomb, tbuf, M);
  merge_kernel<<<R, 256, 0, stream>>>(feat_f0, feat_f1, b_ids, i_ids, j_ids,
                                      stride_p, bpackM, tbuf, out, M);
}
